// GNNStack_86466281603777
// MI455X (gfx1250) — compile-verified, matrix operand order checked by reading, not run on hardware
//
#include <hip/hip_runtime.h>
#include <hip/hip_bf16.h>

typedef __attribute__((ext_vector_type(16))) _Float16 v16h;
typedef __attribute__((ext_vector_type(8)))  float    v8f;

#define HC   64      // H*C channels
#define NEGS 0.2f    // leaky relu slope
#define DOUT 40

// ---------------------------------------------------------------------------
// Fused node linear: Y = X @ W^T + b ; xl = Y ; al = att_l*Y ; ar = att_r*Y
// X: [N,K] f32 (optionally relu'd on load), W: [64,K] f32 row-major.
// One wave per 16-row panel; 4 WMMA f32_16x16x32_f16 accumulators cover 64 cols.
// ---------------------------------------------------------------------------
template <bool RELU_IN>
__global__ void gat_linear_kernel(const float* __restrict__ X,
                                  const float* __restrict__ W,
                                  const float* __restrict__ bias,
                                  const float* __restrict__ att_l,
                                  const float* __restrict__ att_r,
                                  float* __restrict__ xl,
                                  float* __restrict__ al,
                                  float* __restrict__ ar,
                                  int N, int K)
{
    const int lane  = threadIdx.x;          // wave32
    const int m     = lane & 15;
    const int kbase = (lane >> 4) * 8;      // 0 or 8
    const int row0  = blockIdx.x * 16;

    int arow = row0 + m;
    if (arow >= N) arow = N - 1;            // clamp (N is multiple of 16 anyway)

    v8f acc0 = {}, acc1 = {}, acc2 = {}, acc3 = {};

    for (int k0 = 0; k0 < K; k0 += 32) {
        // ---- A fragment (convert f32 -> f16, optional relu) ----
        v16h a;
        const float* xrow = X + (size_t)arow * K + k0 + kbase;
#pragma unroll
        for (int i = 0; i < 8; ++i) {
            float v0 = xrow[i];
            float v1 = xrow[16 + i];
            if (RELU_IN) { v0 = fmaxf(v0, 0.f); v1 = fmaxf(v1, 0.f); }
            a[i]     = (_Float16)v0;
            a[8 + i] = (_Float16)v1;
        }
        // ---- B fragments: B[k][n] = W[n][k] ----
#pragma unroll
        for (int t = 0; t < 4; ++t) {
            const int n = t * 16 + m;
            const float* wrow = W + (size_t)n * K + k0 + kbase;
            v16h b;
#pragma unroll
            for (int i = 0; i < 8; ++i) {
                b[i]     = (_Float16)wrow[i];
                b[8 + i] = (_Float16)wrow[16 + i];
            }
            v8f* accp = (t == 0) ? &acc0 : (t == 1) ? &acc1 : (t == 2) ? &acc2 : &acc3;
            *accp = __builtin_amdgcn_wmma_f32_16x16x32_f16(
                false, a, false, b, (short)0, *accp, false, false);
        }
    }

    // ---- epilogue: bias + attention premultiplies ----
    const int rowadd = (lane >> 4) * 8;
#pragma unroll
    for (int t = 0; t < 4; ++t) {
        const int col  = t * 16 + m;
        const float bb  = bias[col];
        const float atl = att_l[col];
        const float atr = att_r[col];
        const v8f acc = (t == 0) ? acc0 : (t == 1) ? acc1 : (t == 2) ? acc2 : acc3;
#pragma unroll
        for (int r = 0; r < 8; ++r) {
            const int row = row0 + rowadd + r;
            if (row < N) {
                const float  y   = acc[r] + bb;
                const size_t idx = (size_t)row * HC + col;
                xl[idx] = y;
                al[idx] = atl * y;
                ar[idx] = atr * y;
            }
        }
    }
}

// ---------------------------------------------------------------------------
// Edge passes, vectorized: thread = edge*16 + channel-group (4 ch per thread).
// Each thread does 128-bit gathers from the L2-resident [N,64] tables.
// ---------------------------------------------------------------------------
__device__ __forceinline__ unsigned int ord_encode(float v) {
    unsigned int b = __float_as_uint(v);
    return (b & 0x80000000u) ? ~b : (b | 0x80000000u);
}
__device__ __forceinline__ float ord_decode_guard(unsigned int u) {
    if (u == 0u) return 0.f;                       // untouched segment -> m = 0
    unsigned int b = (u & 0x80000000u) ? (u ^ 0x80000000u) : ~u;
    return __uint_as_float(b);
}
__device__ __forceinline__ float leaky(float v) {
    return (v >= 0.f) ? v : NEGS * v;
}

__global__ void edge_max_kernel(const int* __restrict__ src,
                                const int* __restrict__ dst,
                                const float* __restrict__ al,
                                const float* __restrict__ ar,
                                unsigned int* __restrict__ mord,
                                long long total)   // total = E*16
{
    long long t = (long long)blockIdx.x * blockDim.x + threadIdx.x;
    if (t >= total) return;
    const long long e = t >> 4;
    const int c = (int)(t & 15) * 4;
    const int s = src[e], d = dst[e];
    const size_t di = (size_t)d * HC + c;
    const float4 a4 = *(const float4*)(al + (size_t)s * HC + c);
    const float4 r4 = *(const float4*)(ar + di);
    unsigned int* mp = mord + di;
    atomicMax(mp + 0, ord_encode(leaky(r4.x + a4.x)));
    atomicMax(mp + 1, ord_encode(leaky(r4.y + a4.y)));
    atomicMax(mp + 2, ord_encode(leaky(r4.z + a4.z)));
    atomicMax(mp + 3, ord_encode(leaky(r4.w + a4.w)));
}

__global__ void edge_expsum_kernel(const int* __restrict__ src,
                                   const int* __restrict__ dst,
                                   const float* __restrict__ al,
                                   const float* __restrict__ ar,
                                   const unsigned int* __restrict__ mord,
                                   float* __restrict__ ssum,
                                   long long total)
{
    long long t = (long long)blockIdx.x * blockDim.x + threadIdx.x;
    if (t >= total) return;
    const long long e = t >> 4;
    const int c = (int)(t & 15) * 4;
    const int s = src[e], d = dst[e];
    const size_t di = (size_t)d * HC + c;
    const float4 a4 = *(const float4*)(al + (size_t)s * HC + c);
    const float4 r4 = *(const float4*)(ar + di);
    const uint4  m4 = *(const uint4*)(mord + di);
    float* sp = ssum + di;
    atomicAdd(sp + 0, __expf(leaky(r4.x + a4.x) - ord_decode_guard(m4.x)));
    atomicAdd(sp + 1, __expf(leaky(r4.y + a4.y) - ord_decode_guard(m4.y)));
    atomicAdd(sp + 2, __expf(leaky(r4.z + a4.z) - ord_decode_guard(m4.z)));
    atomicAdd(sp + 3, __expf(leaky(r4.w + a4.w) - ord_decode_guard(m4.w)));
}

__global__ void edge_aggregate_kernel(const int* __restrict__ src,
                                      const int* __restrict__ dst,
                                      const float* __restrict__ al,
                                      const float* __restrict__ ar,
                                      const float* __restrict__ xl,
                                      const unsigned int* __restrict__ mord,
                                      const float* __restrict__ ssum,
                                      float* __restrict__ agg,
                                      long long total)
{
    long long t = (long long)blockIdx.x * blockDim.x + threadIdx.x;
    if (t >= total) return;
    const long long e = t >> 4;
    const int c = (int)(t & 15) * 4;
    const int s = src[e], d = dst[e];
    const size_t di = (size_t)d * HC + c;
    const size_t si = (size_t)s * HC + c;
    const float4 a4 = *(const float4*)(al + si);
    const float4 r4 = *(const float4*)(ar + di);
    const uint4  m4 = *(const uint4*)(mord + di);
    const float4 s4 = *(const float4*)(ssum + di);
    const float4 x4 = *(const float4*)(xl + si);
    float* gp = agg + di;
    atomicAdd(gp + 0, x4.x * (__expf(leaky(r4.x + a4.x) - ord_decode_guard(m4.x)) / (s4.x + 1e-16f)));
    atomicAdd(gp + 1, x4.y * (__expf(leaky(r4.y + a4.y) - ord_decode_guard(m4.y)) / (s4.y + 1e-16f)));
    atomicAdd(gp + 2, x4.z * (__expf(leaky(r4.z + a4.z) - ord_decode_guard(m4.z)) / (s4.z + 1e-16f)));
    atomicAdd(gp + 3, x4.w * (__expf(leaky(r4.w + a4.w) - ord_decode_guard(m4.w)) / (s4.w + 1e-16f)));
}

// ---------------------------------------------------------------------------
// Post-MP head: relu -> [64x32] -> [32x40] -> log_softmax. One block per node.
// ---------------------------------------------------------------------------
__global__ void post_kernel(const float* __restrict__ agg,
                            const float* __restrict__ Wp1,
                            const float* __restrict__ bp1,
                            const float* __restrict__ Wp2,
                            const float* __restrict__ bp2,
                            float* __restrict__ out, int N)
{
    __shared__ float h[64];
    __shared__ float tt[32];
    __shared__ float o[DOUT];
    __shared__ float red[64];

    const int node = blockIdx.x;
    const int tid  = threadIdx.x;
    if (node >= N) return;

    h[tid] = fmaxf(agg[(size_t)node * HC + tid], 0.f);
    __syncthreads();

    if (tid < 32) {
        float acc = bp1[tid];
        const float* w = Wp1 + tid * 64;
#pragma unroll
        for (int k = 0; k < 64; ++k) acc += h[k] * w[k];
        tt[tid] = acc;
    }
    __syncthreads();

    if (tid < DOUT) {
        float acc = bp2[tid];
        const float* w = Wp2 + tid * 32;
#pragma unroll
        for (int k = 0; k < 32; ++k) acc += tt[k] * w[k];
        o[tid] = acc;
    }
    __syncthreads();

    red[tid] = (tid < DOUT) ? o[tid] : -INFINITY;
    __syncthreads();
    for (int s = 32; s > 0; s >>= 1) {
        if (tid < s) red[tid] = fmaxf(red[tid], red[tid + s]);
        __syncthreads();
    }
    const float mx = red[0];
    __syncthreads();
    red[tid] = (tid < DOUT) ? __expf(o[tid] - mx) : 0.f;
    __syncthreads();
    for (int s = 32; s > 0; s >>= 1) {
        if (tid < s) red[tid] += red[tid + s];
        __syncthreads();
    }
    const float lse = __logf(red[0]) + mx;
    if (tid < DOUT) out[(size_t)node * DOUT + tid] = o[tid] - lse;
}

// ---------------------------------------------------------------------------
extern "C" void kernel_launch(void* const* d_in, const int* in_sizes, int n_in,
                              void* d_out, int out_size, void* d_ws, size_t ws_size,
                              hipStream_t stream) {
    const float* x       = (const float*)d_in[0];
    const int*   ei      = (const int*)d_in[1];
    const float* W1      = (const float*)d_in[2];
    const float* b1      = (const float*)d_in[3];
    const float* att_l1  = (const float*)d_in[4];
    const float* att_r1  = (const float*)d_in[5];
    const float* W2      = (const float*)d_in[6];
    const float* b2      = (const float*)d_in[7];
    const float* att_l2  = (const float*)d_in[8];
    const float* att_r2  = (const float*)d_in[9];
    const float* Wp1     = (const float*)d_in[10];
    const float* bp1     = (const float*)d_in[11];
    const float* Wp2     = (const float*)d_in[12];
    const float* bp2     = (const float*)d_in[13];
    float*       out     = (float*)d_out;

    const int N = in_sizes[0] / 128;       // 100000
    const int E = in_sizes[1] / 2;         // 1600000
    const int* src = ei;
    const int* dst = ei + E;

    const size_t n64 = (size_t)N * HC;
    float*        xl   = (float*)d_ws;
    float*        al   = xl + n64;
    float*        ar   = al + n64;
    unsigned int* mord = (unsigned int*)(ar + n64);
    float*        ssum = (float*)(mord + n64);
    float*        agg  = ssum + n64;

    const long long total   = (long long)E * 16;   // 4 channels per thread
    const int       eblocks = (int)((total + 255) / 256);
    const int       gblocks = (N + 15) / 16;

    // ---------------- layer 1 ----------------
    gat_linear_kernel<false><<<gblocks, 32, 0, stream>>>(
        x, W1, b1, att_l1, att_r1, xl, al, ar, N, 128);
    hipMemsetAsync(mord, 0, 3 * n64 * sizeof(float), stream);   // mord, ssum, agg
    edge_max_kernel<<<eblocks, 256, 0, stream>>>(src, dst, al, ar, mord, total);
    edge_expsum_kernel<<<eblocks, 256, 0, stream>>>(src, dst, al, ar, mord, ssum, total);
    edge_aggregate_kernel<<<eblocks, 256, 0, stream>>>(src, dst, al, ar, xl, mord, ssum,
                                                       agg, total);

    // ---------------- layer 2 (relu fused into A-fragment load) ----------------
    gat_linear_kernel<true><<<gblocks, 32, 0, stream>>>(
        agg, W2, b2, att_l2, att_r2, xl, al, ar, N, 64);
    hipMemsetAsync(mord, 0, 2 * n64 * sizeof(float), stream);   // mord, ssum
    hipMemsetAsync(agg, 0, n64 * sizeof(float), stream);
    edge_max_kernel<<<eblocks, 256, 0, stream>>>(src, dst, al, ar, mord, total);
    edge_expsum_kernel<<<eblocks, 256, 0, stream>>>(src, dst, al, ar, mord, ssum, total);
    edge_aggregate_kernel<<<eblocks, 256, 0, stream>>>(src, dst, al, ar, xl, mord, ssum,
                                                       agg, total);

    // ---------------- MLP head + log_softmax ----------------
    post_kernel<<<N, 64, 0, stream>>>(agg, Wp1, bp1, Wp2, bp2, out, N);
}
